// DegModel_47871705481651
// MI455X (gfx1250) — matrix operand order repaired and checked by reading, not compiled, not run on hardware
//
#include <hip/hip_runtime.h>

typedef __attribute__((ext_vector_type(16))) __bf16 v16bf;
typedef __attribute__((ext_vector_type(8)))  float  v8f;

#define NPIX_TOTAL 65536   // B*Hl*Wl = 16*64*64
#define WAVES      8
#define W_IN_OFF   0
#define W1_OFF     4096
#define W2_OFF     36864
#define WOUT_OFF   69632
#define WS_ELEMS   98304   // 69632 + 448*64

__device__ __forceinline__ v8f wmma_bf16(v16bf a, v16bf b, v8f c) {
  return __builtin_amdgcn_wmma_f32_16x16x32_bf16(false, a, false, b, (short)0, c,
                                                 false, false);
}

// A tile (16xK=32 bf16) for output-row tile mt, K-tile kt, from row-major W[64..448][64]
__device__ __forceinline__ v16bf load_A(const __bf16* __restrict__ W, int mt, int kt,
                                        int lane) {
  v16bf a;
  int M  = mt * 16 + (lane & 15);
  int hi = (lane >> 4) & 1;
  const __bf16* row = W + M * 64;
#pragma unroll
  for (int v = 0; v < 8; ++v) {
    int K = kt * 32 + hi * 16 + v * 2;
    a[2 * v]     = row[K];
    a[2 * v + 1] = row[K + 1];
  }
  return a;
}

// B tile (K=32 x N=16) from per-wave LDS h staging, layout [16 pix][64 ch]
__device__ __forceinline__ v16bf load_B_lds(const __bf16* __restrict__ L, int k2,
                                            int lane) {
  v16bf b;
  int col = lane & 15;
  int hi  = (lane >> 4) & 1;
  const __bf16* p = L + col * 64 + k2 * 32 + hi * 16;
#pragma unroll
  for (int j = 0; j < 16; ++j) b[j] = p[j];
  return b;
}

// scatter a 16x16 f32 C tile into [pix][ch] bf16 LDS staging (optional relu)
__device__ __forceinline__ void store_C_lds(__bf16* __restrict__ L, v8f c, int mt,
                                            int lane, bool relu) {
  int col = lane & 15;
  int hi  = (lane >> 4) & 1;
#pragma unroll
  for (int g = 0; g < 8; ++g) {
    int row = mt * 16 + g + hi * 8;
    float v = c[g];
    if (relu && v < 0.f) v = 0.f;
    L[col * 64 + row] = (__bf16)v;
  }
}

__global__ __launch_bounds__(256) void convert_weights_kernel(
    const float* __restrict__ w_in, const float* __restrict__ w1,
    const float* __restrict__ w2, const float* __restrict__ wo,
    __bf16* __restrict__ ws) {
  int i = blockIdx.x * blockDim.x + threadIdx.x;
  if (i >= WS_ELEMS) return;
  if (i < W1_OFF)            ws[i] = (__bf16)w_in[i];
  else if (i < W2_OFF)       ws[i] = (__bf16)w1[i - W1_OFF];
  else if (i < WOUT_OFF)     ws[i] = (__bf16)w2[i - W2_OFF];
  else {
    int j = i - WOUT_OFF;   // padded 448x64, rows >= 441 are zero
    ws[i] = (j < 441 * 64) ? (__bf16)wo[j] : (__bf16)0.0f;
  }
}

__global__ __launch_bounds__(256) void trunk_kernel(
    const float* __restrict__ z, const __bf16* __restrict__ ws,
    const float* __restrict__ b_in, const float* __restrict__ rb1,
    const float* __restrict__ rb2, const float* __restrict__ b_out,
    float* __restrict__ kern) {
  __shared__ __bf16 sH[WAVES][16 * 64];
  __shared__ __bf16 sR[WAVES][16 * 64];

  const int lane = threadIdx.x & 31;
  const int wave = threadIdx.x >> 5;
  const int col  = lane & 15;
  const int hi   = (lane >> 4) & 1;
  const int pix0 = (blockIdx.x * WAVES + wave) * 16;  // 16 consecutive hw, same image
  const int b    = pix0 >> 12;
  const int hw0  = pix0 & 4095;

  __bf16* myH = sH[wave];
  __bf16* myR = sR[wave];
  const __bf16* Win = ws + W_IN_OFF;
  const __bf16* W1b = ws + W1_OFF;
  const __bf16* W2b = ws + W2_OFF;
  const __bf16* Wo  = ws + WOUT_OFF;

  // ---- load z B-fragments (K=64 -> two K-tiles), convert f32 -> bf16 ----
  const float* zp = z + (size_t)b * 64 * 4096 + hw0 + col;
  v16bf bz0, bz1;
#pragma unroll
  for (int j = 0; j < 16; ++j) {
    int ch = hi * 16 + j;
    bz0[j] = (__bf16)zp[(size_t)ch * 4096];
    bz1[j] = (__bf16)zp[(size_t)(32 + ch) * 4096];
  }

  // ---- h = W_in * z + b_in ----
  v8f hc[4];
#pragma unroll
  for (int mt = 0; mt < 4; ++mt) {
    v8f c = {};
    c = wmma_bf16(load_A(Win, mt, 0, lane), bz0, c);
    c = wmma_bf16(load_A(Win, mt, 1, lane), bz1, c);
#pragma unroll
    for (int g = 0; g < 8; ++g) c[g] += b_in[mt * 16 + g + hi * 8];
    hc[mt] = c;
    store_C_lds(myH, c, mt, lane, false);
  }
  asm volatile("s_wait_dscnt 0" ::: "memory");

  // ---- 8 residual blocks ----
  for (int blk = 0; blk < 8; ++blk) {
    const __bf16* w1 = W1b + blk * 4096;
    const __bf16* w2 = W2b + blk * 4096;
    const float* bb1 = rb1 + blk * 64;
    const float* bb2 = rb2 + blk * 64;

    v16bf bh0 = load_B_lds(myH, 0, lane);
    v16bf bh1 = load_B_lds(myH, 1, lane);
#pragma unroll
    for (int mt = 0; mt < 4; ++mt) {
      v8f c = {};
      c = wmma_bf16(load_A(w1, mt, 0, lane), bh0, c);
      c = wmma_bf16(load_A(w1, mt, 1, lane), bh1, c);
#pragma unroll
      for (int g = 0; g < 8; ++g) c[g] += bb1[mt * 16 + g + hi * 8];
      store_C_lds(myR, c, mt, lane, true);  // relu
    }
    asm volatile("s_wait_dscnt 0" ::: "memory");

    v16bf br0 = load_B_lds(myR, 0, lane);
    v16bf br1 = load_B_lds(myR, 1, lane);
#pragma unroll
    for (int mt = 0; mt < 4; ++mt) {
      v8f c = {};
      c = wmma_bf16(load_A(w2, mt, 0, lane), br0, c);
      c = wmma_bf16(load_A(w2, mt, 1, lane), br1, c);
#pragma unroll
      for (int g = 0; g < 8; ++g) hc[mt][g] += c[g] + bb2[mt * 16 + g + hi * 8];
      store_C_lds(myH, hc[mt], mt, lane, false);
    }
    asm volatile("s_wait_dscnt 0" ::: "memory");
  }

  // ---- logits = W_out(441x64, padded 448) * h + b_out; online softmax ----
  v16bf bh0 = load_B_lds(myH, 0, lane);
  v16bf bh1 = load_B_lds(myH, 1, lane);
  float m = -__builtin_inff(), s = 0.f;
  for (int mt = 0; mt < 28; ++mt) {
    v8f c = {};
    c = wmma_bf16(load_A(Wo, mt, 0, lane), bh0, c);
    c = wmma_bf16(load_A(Wo, mt, 1, lane), bh1, c);
#pragma unroll
    for (int g = 0; g < 8; ++g) {
      int row = mt * 16 + g + hi * 8;
      if (row < 441) {
        float v  = c[g] + b_out[row];
        float nm = fmaxf(m, v);
        s = s * __expf(m - nm) + __expf(v - nm);
        m = nm;
      }
    }
  }
  // merge the two lane-halves holding the same pixel column
  float om = __shfl_xor(m, 16, 32);
  float os = __shfl_xor(s, 16, 32);
  float M  = fmaxf(m, om);
  float S  = s * __expf(m - M) + os * __expf(om - M);
  float rS = 1.0f / S;

  // ---- pass 2: recompute logits, write softmax, coalesced over hw ----
  float* kp = kern + (size_t)b * 441 * 4096 + hw0 + col;
  for (int mt = 0; mt < 28; ++mt) {
    v8f c = {};
    c = wmma_bf16(load_A(Wo, mt, 0, lane), bh0, c);
    c = wmma_bf16(load_A(Wo, mt, 1, lane), bh1, c);
#pragma unroll
    for (int g = 0; g < 8; ++g) {
      int row = mt * 16 + g + hi * 8;
      if (row < 441) {
        float v = c[g] + b_out[row];
        kp[(size_t)row * 4096] = __expf(v - M) * rS;
      }
    }
  }
}

__device__ __forceinline__ int refl(int i) {
  i = (i < 0) ? -i : i;
  return (i > 255) ? (510 - i) : i;   // pad=10 <= 255 -> single reflection
}

__global__ __launch_bounds__(256) void apply_kernel(
    const float* __restrict__ x, const float* __restrict__ kern,
    float* __restrict__ out) {
  int p = blockIdx.x * blockDim.x + threadIdx.x;  // one thread per (b, hw)
  if (p >= NPIX_TOTAL) return;
  int b  = p >> 12, hw = p & 4095;
  int yl = hw >> 6, xl = hw & 63;

  int rx[21], ry[21];
#pragma unroll
  for (int t = 0; t < 21; ++t) {
    rx[t] = refl(xl * 4 + t - 10);
    ry[t] = refl(yl * 4 + t - 10);
  }

  const float* xb = x + (size_t)b * 3 * 65536;
  const float* kb = kern + (size_t)b * 441 * 4096 + hw;
  float a0 = 0.f, a1 = 0.f, a2 = 0.f;
  for (int dy = 0; dy < 21; ++dy) {
    const float* row = xb + ry[dy] * 256;
#pragma unroll
    for (int dx = 0; dx < 21; ++dx) {
      float kv = kb[(size_t)(dy * 21 + dx) * 4096];
      int ix = rx[dx];
      a0 += kv * row[ix];
      a1 += kv * row[65536 + ix];
      a2 += kv * row[131072 + ix];
    }
  }
  float* ob = out + (size_t)b * 3 * 4096 + hw;
  ob[0]      = a0;
  ob[4096]   = a1;
  ob[8192]   = a2;
}

extern "C" void kernel_launch(void* const* d_in, const int* in_sizes, int n_in,
                              void* d_out, int out_size, void* d_ws, size_t ws_size,
                              hipStream_t stream) {
  const float* x     = (const float*)d_in[0];
  const float* z     = (const float*)d_in[1];
  const float* w_in  = (const float*)d_in[2];
  const float* b_in  = (const float*)d_in[3];
  const float* rw1   = (const float*)d_in[4];
  const float* rb1   = (const float*)d_in[5];
  const float* rw2   = (const float*)d_in[6];
  const float* rb2   = (const float*)d_in[7];
  const float* w_out = (const float*)d_in[8];
  const float* b_out = (const float*)d_in[9];

  float* out  = (float*)d_out;                 // (16,3,64,64)
  float* kern = out + 16 * 3 * 64 * 64;        // (16,441,64,64)
  __bf16* ws  = (__bf16*)d_ws;                 // bf16 weights, 196608 bytes

  convert_weights_kernel<<<(WS_ELEMS + 255) / 256, 256, 0, stream>>>(
      w_in, rw1, rw2, w_out, ws);
  trunk_kernel<<<NPIX_TOTAL / (WAVES * 16), 256, 0, stream>>>(
      z, ws, b_in, rb1, rb2, b_out, kern);
  apply_kernel<<<NPIX_TOTAL / 256, 256, 0, stream>>>(x, kern, out);
}
